// LSHDecoder_1529008357807
// MI455X (gfx1250) — matrix operand order repaired and checked by reading, not compiled, not run on hardware
//
#include <hip/hip_runtime.h>
#include <math.h>

// ---------------------------------------------------------------------------
// LSH decoder, MI455X (gfx1250, wave32, WMMA + async-to-LDS).
//
// Shapes: N=8192 (derived), D=256, 128 planes (16 bands x 8 rows).
// out[i][j] = 1.0f iff (exists band b: 8 sign bits equal) AND cos(Z_i,Z_j)>0.5
//             AND i != j, else 0.0f.
//
// Key identity: with proj sign bits packed row-major into a [128][N/8-byte]
// bit matrix, the 16 signature bytes of element i are the 16 consecutive
// bytes at flat offset i*16.  So the bit matrix IS Sig[N][16 bytes].
//
// Roofline: output write = 256MB ~ 11us @ 23.3TB/s; Gram = 34 GFLOP -> bf16
// v_wmma_f32_16x16x32 puts compute at/below the store bound.  Tile staging
// uses global_load_async_to_lds_b128 (ASYNCcnt) to skip the VGPR round-trip.
// ---------------------------------------------------------------------------

typedef __attribute__((ext_vector_type(16))) __bf16    v16bf;
typedef __attribute__((ext_vector_type(8)))  float     v8f;
typedef __attribute__((ext_vector_type(4)))  float     v4f;
typedef __attribute__((ext_vector_type(4)))  unsigned  v4u;

static __device__ __forceinline__ unsigned f2bf1(float f) {
  unsigned u = __float_as_uint(f);
  return (u + 0x7FFFu + ((u >> 16) & 1u)) >> 16;   // round-to-nearest-even
}
static __device__ __forceinline__ unsigned packbf(float a, float b) {
  return f2bf1(a) | (f2bf1(b) << 16);
}
static __device__ __forceinline__ v16bf mkfrag(v4u lo, v4u hi) {
  union { struct { v4u a, b; } u; v16bf v; } t;
  t.u.a = lo; t.u.b = hi;
  return t.v;
}
// any byte of v is zero?
static __device__ __forceinline__ unsigned hzb(unsigned v) {
  return (v - 0x01010101u) & ~v & 0x80808080u;
}

// async copy of 64B (global, 16B-aligned) -> LDS; tracked by ASYNCcnt.
// INST_OFFSET applies to both global and LDS addresses (ISA 15.18.3 / 8.4.4).
static __device__ __forceinline__ void async_copy64(const v4u* src, v4u* dstLds) {
  unsigned lds = (unsigned)(size_t)dstLds;                 // DS-relative addr
  unsigned long long ga = (unsigned long long)(size_t)src; // 64-bit global
  asm volatile("global_load_async_to_lds_b128 %0, %1, off\n\t"
               "global_load_async_to_lds_b128 %0, %1, off offset:16\n\t"
               "global_load_async_to_lds_b128 %0, %1, off offset:32\n\t"
               "global_load_async_to_lds_b128 %0, %1, off offset:48"
               :: "v"(lds), "v"(ga) : "memory");
}
static __device__ __forceinline__ void wait_async0() {
  asm volatile("s_wait_asynccnt 0x0" ::: "memory");
}

// ---------------- kernel 1: f32 -> bf16 (RNE), 8 elems/thread ---------------
__global__ __launch_bounds__(256) void k_convert(const float* __restrict__ X,
                                                 v4u* __restrict__ Xbf4, int n8) {
  int t = blockIdx.x * blockDim.x + threadIdx.x;
  if (t >= n8) return;
  const v4f* x4 = (const v4f*)X;
  v4f f0 = x4[2 * t], f1 = x4[2 * t + 1];
  v4u o = { packbf(f0.x, f0.y), packbf(f0.z, f0.w),
            packbf(f1.x, f1.y), packbf(f1.z, f1.w) };
  Xbf4[t] = o;
}

// ---------------- kernel 2: rn[i] = 1 / max(||Z_i||, 1e-8) ------------------
__global__ __launch_bounds__(256) void k_rnorm(const float* __restrict__ Z,
                                               float* __restrict__ rn, int N) {
  int wave = threadIdx.x >> 5, lane = threadIdx.x & 31;
  int row  = blockIdx.x * 8 + wave;
  const v4f* z4 = (const v4f*)(Z + (size_t)row * 256);
  v4f a = z4[2 * lane], b = z4[2 * lane + 1];
  float s = a.x*a.x + a.y*a.y + a.z*a.z + a.w*a.w
          + b.x*b.x + b.y*b.y + b.z*b.z + b.w*b.w;
  #pragma unroll
  for (int m = 16; m; m >>= 1) s += __shfl_xor(s, m, 32);
  if (lane == 0) rn[row] = 1.0f / fmaxf(sqrtf(s), 1e-8f);
}

// ---------------- kernel 3: proj signs -> packed signature bits -------------
// grid: N/16 blocks x 256 threads (8 waves). Wave w computes the 16x16 proj
// tile (planes rows w*16..w*16+15) x (Z cols n0..n0+15) via WMMA over K=256,
// then packs sign bits cross-lane and stores two u16 per accumulator VGPR.
__global__ __launch_bounds__(256) void k_sig(const v4u* __restrict__ Zbf4,
                                             const v4u* __restrict__ Pbf4,
                                             unsigned short* __restrict__ Sig16,
                                             int N) {
  int wave = threadIdx.x >> 5, lane = threadIdx.x & 31;
  int hf = lane >> 4, ln = lane & 15;
  int pbase = wave * 16;
  int n0 = blockIdx.x * 16;
  v8f c = {};
  #pragma unroll
  for (int k0 = 0; k0 < 256; k0 += 32) {
    // A fragment (16x32 bf16): row pbase+ln; lane<16 chunks {0,1} of the
    // slab (K k0..k0+15), lane>=16 chunks {2,3} (K k0+16..k0+31).
    const v4u* pr = Pbf4 + (size_t)(pbase + ln) * 32 + (k0 >> 3);
    v4u alo = pr[hf ? 2 : 0];
    v4u ahi = pr[hf ? 3 : 1];
    // B fragment (32x16): column n0+ln = Z row n0+ln; lane<16 K k0..k0+15,
    // lane>=16 K k0+16..k0+31 (contiguous in Zbf).
    const v4u* zr = Zbf4 + (size_t)(n0 + ln) * 32 + ((k0 + hf * 16) >> 3);
    c = __builtin_amdgcn_wmma_f32_16x16x32_bf16(false, mkfrag(alo, ahi),
                                                false, mkfrag(zr[0], zr[1]),
                                                (short)0, c, false, false);
  }
  // C layout: VGPR v -> lanes 0-15: row pbase+v, lanes 16-31: row pbase+v+8,
  // col = n0 + (lane&15). Cross-lane OR-reduce builds the 32-bit sign mask.
  int colu16 = N >> 4;
  #pragma unroll
  for (int v = 0; v < 8; ++v) {
    unsigned bit = (c[v] >= 0.0f) ? (1u << lane) : 0u;
    #pragma unroll
    for (int m = 16; m; m >>= 1) bit |= __shfl_xor(bit, m, 32);
    if (lane == 0) {
      Sig16[(pbase + v)     * colu16 + (n0 >> 4)] = (unsigned short)(bit & 0xFFFFu);
      Sig16[(pbase + v + 8) * colu16 + (n0 >> 4)] = (unsigned short)(bit >> 16);
    }
  }
}

// ---------------- kernel 4: fused Gram + cosine + signature + mask ----------
// 128x128 output tile per block; 8 waves in a 4(M)x2(N) grid; each wave owns
// a 32x64 region = 2x4 accumulators of 16x16. Double-buffered LDS tiles
// filled by global_load_async_to_lds_b128; rows padded to 5 x v4u (80B) so
// the per-lane ds_read_b128 fragment gathers hit distinct bank quads.
__global__ __launch_bounds__(256) void k_pairs(const v4u* __restrict__ Zbf4,
                                               const v4u* __restrict__ Sig4,
                                               const float* __restrict__ rn,
                                               float* __restrict__ out, int N) {
  __shared__ v4u AS[2][128 * 5];
  __shared__ v4u BS[2][128 * 5];
  __shared__ v4u sgA[128], sgB[128];
  __shared__ float rnA[128], rnB[128];

  const int t  = threadIdx.x;
  const int i0 = blockIdx.y * 128, j0 = blockIdx.x * 128;

  if (t < 128) { sgA[t] = Sig4[i0 + t]; rnA[t] = rn[i0 + t]; }
  else         { int r = t - 128; sgB[r] = Sig4[j0 + r]; rnB[r] = rn[j0 + r]; }

  // staging: thread t owns one 64B row-chunk (32 bf16 of the K-slab) of tile
  // A (t<128) or tile B (t>=128), copied global->LDS asynchronously.
  const int lr   = t & 127;
  const int lsel = t >> 7;
  const size_t srow = (size_t)((lsel ? j0 : i0) + lr) * 32;  // 32 v4u per Z row

  async_copy64(Zbf4 + srow, (lsel ? BS[0] : AS[0]) + lr * 5);
  wait_async0();
  __syncthreads();

  const int wave = t >> 5, lane = t & 31;
  const int hf = lane >> 4, ln = lane & 15;
  const int wm = wave & 3, wn = wave >> 2;

  v8f zero = {};
  v8f acc[2][4];
  #pragma unroll
  for (int i = 0; i < 2; ++i)
    #pragma unroll
    for (int j = 0; j < 4; ++j) acc[i][j] = zero;

  int buf = 0;
  #pragma unroll
  for (int step = 0; step < 8; ++step) {            // K = 256 in slabs of 32
    if (step < 7)                                   // async-prefetch next slab
      async_copy64(Zbf4 + srow + (size_t)(step + 1) * 4,
                   (lsel ? BS[buf ^ 1] : AS[buf ^ 1]) + lr * 5);

    v16bf fa[2], fb[4];
    #pragma unroll
    for (int i = 0; i < 2; ++i) {                   // A: chunks {0,1}/{2,3}
      int row = wm * 32 + i * 16 + ln;
      fa[i] = mkfrag(AS[buf][row * 5 + (hf ? 2 : 0)],
                     AS[buf][row * 5 + (hf ? 3 : 1)]);
    }
    #pragma unroll
    for (int j = 0; j < 4; ++j) {                   // B: chunks {0,1}/{2,3}
      int row = wn * 64 + j * 16 + ln;
      fb[j] = mkfrag(BS[buf][row * 5 + (hf ? 2 : 0)],
                     BS[buf][row * 5 + (hf ? 3 : 1)]);
    }
    #pragma unroll
    for (int i = 0; i < 2; ++i)
      #pragma unroll
      for (int j = 0; j < 4; ++j)
        acc[i][j] = __builtin_amdgcn_wmma_f32_16x16x32_bf16(
            false, fa[i], false, fb[j], (short)0, acc[i][j], false, false);

    wait_async0();
    __syncthreads();
    buf ^= 1;
  }

  // fused epilogue: cos = G * rn_i * rn_j > 0.5; band match = any of the 16
  // signature bytes equal (per-dword XOR + haszero); kill the diagonal.
  v4u sb[4]; float rb[4]; int njl[4];
  #pragma unroll
  for (int j = 0; j < 4; ++j) {
    njl[j] = wn * 64 + j * 16 + ln;
    sb[j] = sgB[njl[j]]; rb[j] = rnB[njl[j]];
  }
  #pragma unroll
  for (int i = 0; i < 2; ++i) {
    #pragma unroll
    for (int v = 0; v < 8; ++v) {
      int mi = wm * 32 + i * 16 + hf * 8 + v;       // C layout: M = v (+8 hi half)
      v4u sa = sgA[mi]; float ra = rnA[mi];
      int gi = i0 + mi;
      float* orow = out + (size_t)gi * N + j0;
      #pragma unroll
      for (int j = 0; j < 4; ++j) {
        float g = acc[i][j][v];
        bool cosok = (g * ra * rb[j]) > 0.5f;
        unsigned m = hzb(sa.x ^ sb[j].x) | hzb(sa.y ^ sb[j].y)
                   | hzb(sa.z ^ sb[j].z) | hzb(sa.w ^ sb[j].w);
        bool ok = (m != 0u) & cosok & (gi != (j0 + njl[j]));
        orow[njl[j]] = ok ? 1.0f : 0.0f;
      }
    }
  }
}

// ---------------------------------------------------------------------------
extern "C" void kernel_launch(void* const* d_in, const int* in_sizes, int n_in,
                              void* d_out, int out_size, void* d_ws, size_t ws_size,
                              hipStream_t stream) {
  (void)n_in; (void)out_size; (void)ws_size;
  const float* Z      = (const float*)d_in[0];
  const float* planes = (const float*)d_in[1];
  float* out = (float*)d_out;
  const int N = in_sizes[0] / 256;                  // 8192
  const int P = in_sizes[1] / 256;                  // 128 planes

  // workspace: Zbf (N*512B) | Sig (N*16B) | rnorm (N*4B) | Pbf (P*512B)
  char* ws = (char*)d_ws;
  v4u* Zbf4 = (v4u*)ws;
  size_t zbf_bytes = (size_t)N * 512;
  unsigned short* Sig16 = (unsigned short*)(ws + zbf_bytes);
  v4u* Sig4 = (v4u*)Sig16;
  size_t sig_bytes = (size_t)N * 16;
  float* rnv = (float*)(ws + zbf_bytes + sig_bytes);
  v4u* Pbf4 = (v4u*)(ws + zbf_bytes + sig_bytes + (size_t)N * 4);

  int n8z = N * 32;                                 // N*256/8 convert threads
  int n8p = P * 32;
  k_convert<<<(n8z + 255) / 256, 256, 0, stream>>>(Z, Zbf4, n8z);
  k_convert<<<(n8p + 255) / 256, 256, 0, stream>>>(planes, Pbf4, n8p);
  k_rnorm  <<<N / 8,             256, 0, stream>>>(Z, rnv, N);
  k_sig    <<<N / 16,            256, 0, stream>>>(Zbf4, Pbf4, Sig16, N);
  dim3 grid(N / 128, N / 128);
  k_pairs  <<<grid,              256, 0, stream>>>(Zbf4, Sig4, rnv, out, N);
}
